// CharDecoder_59794534695132
// MI455X (gfx1250) — compile-verified, matrix operand order checked by reading, not run on hardware
//
#include <hip/hip_runtime.h>

typedef _Float16 half_t;
typedef __attribute__((ext_vector_type(16))) _Float16 v16h;
typedef __attribute__((ext_vector_type(8)))  _Float16 v8h;
typedef __attribute__((ext_vector_type(8)))  float    v8f;
typedef __attribute__((ext_vector_type(4)))  unsigned int u32x4;
typedef __attribute__((ext_vector_type(8)))  int          i32x8;
typedef __attribute__((ext_vector_type(4)))  int          i32x4;

#define BWn 4096   // B*W rows
#define Qd  256    // quantized repr dim
#define Hd  512    // hidden dim
#define Ad  128    // alphabet
#define Cd  16     // char steps

// Load a 16-bit WMMA A/B fragment. Per CDNA5 ISA 16-bit layout:
// lane<16 holds K = k0..k0+7 and k0+16..k0+23 ; lane>=16 gets +8 (folded into
// k0 by the caller via kh*8). Two 16-byte loads per lane.
__device__ __forceinline__ v16h load_frag(const half_t* row, int k0) {
  v8h lo = *reinterpret_cast<const v8h*>(row + k0);
  v8h hi = *reinterpret_cast<const v8h*>(row + k0 + 16);
  return __builtin_shufflevector(lo, hi, 0, 1, 2, 3, 4, 5, 6, 7,
                                 8, 9, 10, 11, 12, 13, 14, 15);
}

__device__ __forceinline__ float sigmoidf_(float x) {
  return 1.0f / (1.0f + __expf(-x));
}

__global__ void __launch_bounds__(256)
cvt_f32_f16(const float* __restrict__ s, half_t* __restrict__ d, int n) {
  int i = blockIdx.x * 256 + threadIdx.x;
  if (i < n) d[i] = (half_t)s[i];
}

// One workgroup = 16 sequence rows, 8 waves. Wave w owns hidden units
// [64w, 64w+64) for gates/cell-state, and output cols [16w, 16w+16) for logits.
// Hidden state is double-buffered in LDS as f16 (one barrier per step).
__global__ void __launch_bounds__(256, 1)
lstm_char_decoder(const half_t* __restrict__ xqH,    // [BW, Q] f16 (preconverted)
                  const int*    __restrict__ tchars, // [BW, C]
                  const half_t* __restrict__ WpH,    // [H, Q]  f16
                  const float*  __restrict__ bp,     // [H]
                  const float*  __restrict__ Wih,    // [4H, A] f32 (one-hot gather)
                  const half_t* __restrict__ WhhH,   // [4H, H] f16
                  const float*  __restrict__ bih,    // [4H]
                  const float*  __restrict__ bhh,    // [4H]
                  const half_t* __restrict__ WoH,    // [A, H]  f16
                  const float*  __restrict__ bo,     // [A]
                  float*        __restrict__ out) {  // [BW, C, A]
  __shared__ half_t hsm[2][16][Hd + 8];  // double-buffered hidden state (16B rows)
  __shared__ half_t xsm[16][Qd + 8];     // staged input rows, f16, padded
  __shared__ int    chsm[16][Cd];        // clamped teacher-forcing chars

  const int tid  = threadIdx.x;
  const int wave = tid >> 5;
  const int lane = tid & 31;
  const int lm   = lane & 15;   // col (B/C/D) or row (A) within tile
  const int kh   = lane >> 4;   // K-half selector for A/B fragments
  const int rowBase = blockIdx.x * 16;

  // ---- stage the 16x256 f16 input tile into LDS ----
#if __has_builtin(__builtin_amdgcn_tensor_load_to_lds) && defined(__gfx1250__)
  if (wave == 0) {
    // Tensor Data Mover: 2D tile [16 rows x 256 cols] of 2-byte elements,
    // LDS padding of 4 dwords (16B) after every 128 dwords (one 512B row)
    // to reproduce the (Qd+8)-element row pitch.
    unsigned long long ga = (unsigned long long)(size_t)(xqH + (size_t)rowBase * Qd);
    unsigned int ldsOff = (unsigned int)(size_t)&xsm[0][0];   // LDS byte offset
    u32x4 g0;
    g0.x = 1u;                                               // count=1, user D#
    g0.y = ldsOff;                                           // lds_addr
    g0.z = (unsigned int)ga;                                 // global_addr[31:0]
    g0.w = (unsigned int)((ga >> 32) & 0x01FFFFFFu) | (2u << 30); // addr[56:32], type=2
    i32x8 g1;
    g1[0] = (int)((1u << 16)       // data_size = 2 bytes
                | (1u << 20)       // pad_enable
                | (6u << 22)       // pad_interval: every 128 dwords
                | (3u << 25));     // pad_amount: 4 dwords
    g1[1] = (int)(256u << 16);     // tensor_dim0 = 256 (bits 79:48, low half)
    g1[2] = (int)(4096u << 16);    // tensor_dim0 hi=0 | tensor_dim1 = 4096 (low)
    g1[3] = (int)(256u << 16);     // tensor_dim1 hi=0 | tile_dim0 = 256
    g1[4] = 16;                    // tile_dim1 = 16, tile_dim2 = 0
    g1[5] = 256;                   // tensor_dim0_stride = 256 (low 32)
    g1[6] = 0;
    g1[7] = 0;
    i32x4 gz = {0, 0, 0, 0};
#if __clang_major__ >= 23
    i32x8 gz8 = {0, 0, 0, 0, 0, 0, 0, 0};
    __builtin_amdgcn_tensor_load_to_lds(g0, g1, gz, gz, gz8, 0);
#else
    __builtin_amdgcn_tensor_load_to_lds(g0, g1, gz, gz, 0);
#endif
    __builtin_amdgcn_s_wait_tensorcnt(0);
  }
#else
  for (int i = tid; i < 16 * Qd; i += 256) {
    int r = i >> 8, c = i & (Qd - 1);
    xsm[r][c] = xqH[(size_t)(rowBase + r) * Qd + c];
  }
#endif
  for (int i = tid; i < 16 * Cd; i += 256) {
    int r = i >> 4, t = i & 15;
    int ch = tchars[(size_t)(rowBase + r) * Cd + t];
    if (ch < 0 || ch >= Ad) ch = 0;   // invalid -> one-hot index 0
    chsm[r][t] = ch;
  }
  __syncthreads();

  v8f cst[4];  // cell state, C-layout, 4 tiles of 16 hidden cols per wave

  // ---- h0 = xq @ Wp^T + bp  -> hsm[0] ----
#pragma unroll
  for (int tt = 0; tt < 4; ++tt) {
    const int n = wave * 64 + tt * 16 + lm;     // hidden unit index
    const float b0 = bp[n];
    v8f acc;
#pragma unroll
    for (int j = 0; j < 8; ++j) { acc[j] = b0; cst[tt][j] = 0.0f; }
    const half_t* brow = WpH + (size_t)n * Qd;  // B[k][n] = Wp[n][k]
#pragma unroll
    for (int kt = 0; kt < Qd / 32; ++kt) {
      v16h a = load_frag(&xsm[lm][0], kt * 32 + kh * 8);
      v16h b = load_frag(brow,        kt * 32 + kh * 8);
      acc = __builtin_amdgcn_wmma_f32_16x16x32_f16(false, a, false, b,
                                                   (short)0, acc, false, false);
    }
#pragma unroll
    for (int j = 0; j < 8; ++j)
      hsm[0][j + 8 * kh][wave * 64 + tt * 16 + lm] = (half_t)acc[j];
  }
  __syncthreads();

  // ---- 16 recurrent steps, one barrier per step via LDS double buffering ----
#pragma unroll 1
  for (int t = 0; t < Cd; ++t) {
    const int cur = t & 1, nxt = cur ^ 1;
    const half_t* hrow = &hsm[cur][lm][0];

#pragma unroll
    for (int tt = 0; tt < 4; ++tt) {
      const int hc = wave * 64 + tt * 16;       // hidden col tile base
      v8f g4[4];
#pragma unroll
      for (int g = 0; g < 4; ++g) {
        const int n = g * Hd + hc + lm;         // gate row of W_ih/W_hh
        const float bias = bih[n] + bhh[n];
        v8f acc;
#pragma unroll
        for (int j = 0; j < 8; ++j) {
          const int m  = j + 8 * kh;                        // sequence row in tile
          const int ct = (t == 0) ? 0 : chsm[m][t - 1];     // teacher-forced input
          acc[j] = bias + Wih[(size_t)n * Ad + ct];         // one-hot @ W_ih^T gather
        }
        const half_t* brow = WhhH + (size_t)n * Hd;         // B[k][n] = W_hh[n][k]
#pragma unroll
        for (int kt = 0; kt < 16; ++kt) {
          v16h a = load_frag(hrow, kt * 32 + kh * 8);
          v16h b = load_frag(brow, kt * 32 + kh * 8);
          acc = __builtin_amdgcn_wmma_f32_16x16x32_f16(false, a, false, b,
                                                       (short)0, acc, false, false);
        }
        g4[g] = acc;
      }
      // LSTM cell update for this 16-col hidden tile (all in registers)
#pragma unroll
      for (int j = 0; j < 8; ++j) {
        const float ig = sigmoidf_(g4[0][j]);
        const float fg = sigmoidf_(g4[1][j]);
        const float gg = tanhf(g4[2][j]);
        const float og = sigmoidf_(g4[3][j]);
        const float cn = fg * cst[tt][j] + ig * gg;
        cst[tt][j] = cn;
        hsm[nxt][j + 8 * kh][hc + lm] = (half_t)(og * tanhf(cn));
      }
    }
    __syncthreads();  // h_{t+1} visible; also fences reads of hsm[cur]

    // ---- logits_t = h_{t+1} @ Wo^T + bo ----
    {
      const int n = wave * 16 + lm;             // output alphabet col
      const float b0 = bo[n];
      v8f acc;
#pragma unroll
      for (int j = 0; j < 8; ++j) acc[j] = b0;
      const half_t* brow = WoH + (size_t)n * Hd;
#pragma unroll
      for (int kt = 0; kt < 16; ++kt) {
        v16h a = load_frag(&hsm[nxt][lm][0], kt * 32 + kh * 8);
        v16h b = load_frag(brow,             kt * 32 + kh * 8);
        acc = __builtin_amdgcn_wmma_f32_16x16x32_f16(false, a, false, b,
                                                     (short)0, acc, false, false);
      }
#pragma unroll
      for (int j = 0; j < 8; ++j) {
        const int r = rowBase + j + 8 * kh;
        out[((size_t)r * Cd + t) * Ad + n] = acc[j];
      }
    }
  }
}

extern "C" void kernel_launch(void* const* d_in, const int* in_sizes, int n_in,
                              void* d_out, int out_size, void* d_ws, size_t ws_size,
                              hipStream_t stream) {
  const float* xq  = (const float*)d_in[0];
  const int*   tch = (const int*)  d_in[1];
  const float* Wp  = (const float*)d_in[2];
  const float* bp  = (const float*)d_in[3];
  const float* Wih = (const float*)d_in[4];
  const float* Whh = (const float*)d_in[5];
  const float* bih = (const float*)d_in[6];
  const float* bhh = (const float*)d_in[7];
  const float* Wo  = (const float*)d_in[8];
  const float* bo  = (const float*)d_in[9];
  float* out = (float*)d_out;

  // f16 copies in workspace (weights stay resident in 192MB L2 across steps)
  half_t* WpH  = (half_t*)d_ws;                       // 512*256
  half_t* WhhH = WpH  + (size_t)Hd * Qd;              // 2048*512
  half_t* WoH  = WhhH + (size_t)4 * Hd * Hd;          // 128*512
  half_t* xqH  = WoH  + (size_t)Ad * Hd;              // 4096*256

  const int nWp = Hd * Qd, nWhh = 4 * Hd * Hd, nWo = Ad * Hd, nXq = BWn * Qd;
  cvt_f32_f16<<<(nWp  + 255) / 256, 256, 0, stream>>>(Wp,  WpH,  nWp);
  cvt_f32_f16<<<(nWhh + 255) / 256, 256, 0, stream>>>(Whh, WhhH, nWhh);
  cvt_f32_f16<<<(nWo  + 255) / 256, 256, 0, stream>>>(Wo,  WoH,  nWo);
  cvt_f32_f16<<<(nXq  + 255) / 256, 256, 0, stream>>>(xq,  xqH,  nXq);

  lstm_char_decoder<<<BWn / 16, 256, 0, stream>>>(xqH, tch, WpH, bp, Wih, WhhH,
                                                  bih, bhh, WoH, bo, out);
}
